// Attention_78683800863453
// MI455X (gfx1250) — matrix operand order repaired and checked by reading, not compiled, not run on hardware
//
#include <hip/hip_runtime.h>
#include <stdint.h>

// ---------------------------------------------------------------------------
// Problem constants (match reference)
// ---------------------------------------------------------------------------
#define BATCH  2
#define N_SEQ  2048
#define DIM    1024
#define HEADS  16
#define DH     64
#define INNER  (HEADS * DH)      // 1024
#define QKV3   (3 * INNER)       // 3072
#define ROWS   (BATCH * N_SEQ)   // 4096
#define LN_EPS 1e-5f

typedef _Float16 half_t;
typedef __attribute__((ext_vector_type(16))) _Float16 v16h;
typedef __attribute__((ext_vector_type(8)))  _Float16 v8h;
typedef __attribute__((ext_vector_type(8)))  float    v8f;
typedef uint32_t u32x4 __attribute__((ext_vector_type(4)));
typedef uint32_t u32x8 __attribute__((ext_vector_type(8)));

union Frag16 { v16h v; v8h o[2]; half_t h[16]; };

__device__ __forceinline__ v8f wmma_f16(v16h a, v16h b, v8f c) {
  // D = A(16x32 f16) * B(32x16 f16) + C(16x16 f32)
  return __builtin_amdgcn_wmma_f32_16x16x32_f16(
      /*neg_a=*/false, a, /*neg_b=*/false, b,
      /*c_mod=*/(short)0, c, /*reuse_a=*/false, /*reuse_b=*/false);
}

// Low 32 bits of a generic pointer to a __shared__ object = wave-relative LDS
// byte offset (flat->LDS discards addr[63:32], per ISA 10.2 aperture rules).
__device__ __forceinline__ uint32_t lds_off(const void* p) {
  return (uint32_t)(uintptr_t)p;
}

// ---------------------------------------------------------------------------
// TDM: DMA a [rows x 64-half] row-major tile (row stride 64 halfs in memory)
// into LDS with +8-half padding per row (LDS row stride 72 halfs = 144B).
// D# layout per CDNA5 ISA ch.8 (2D tensor, data_size=2B, pad 32DW -> +4DW).
// Issue from ONE wave only (wave-uniform branch). Tracked by TENSORcnt.
// ---------------------------------------------------------------------------
__device__ __forceinline__ void tdm_load_rows_f16(uint32_t lds_addr,
                                                  uint64_t gaddr,
                                                  uint32_t tensor_rows,
                                                  uint32_t tile_rows) {
  u32x4 g0;
  g0[0] = 1u;                                            // count=1, user mode
  g0[1] = lds_addr;                                      // lds_addr [63:32]
  g0[2] = (uint32_t)gaddr;                               // global_addr lo
  g0[3] = (uint32_t)((gaddr >> 32) & 0x01FFFFFFu)        // global_addr[56:32]
        | (2u << 30);                                    // type=2 ("image")
  u32x8 g1;
  g1[0] = (1u << 16)                                     // data_size=2B
        | (1u << 20)                                     // pad_enable
        | (4u << 22)                                     // pad_interval: 32 DW
        | (3u << 25);                                    // pad_amount: 4 DW
  g1[1] = (64u << 16);                                   // tensor_dim0=64 (lo16)
  g1[2] = (tensor_rows & 0xFFFFu) << 16;                 // dim0 hi=0 | dim1 lo16
  g1[3] = ((tensor_rows >> 16) & 0xFFFFu) | (64u << 16); // dim1 hi | tile_dim0=64
  g1[4] = tile_rows & 0xFFFFu;                           // tile_dim1 | tile_dim2=0
  g1[5] = 64u;                                           // tensor_dim0_stride lo32
  g1[6] = 0u;                                            // stride hi | dim1_stride lo
  g1[7] = 0u;                                            // dim1_stride hi (unused 2D)
  asm volatile("tensor_load_to_lds %0, %1" :: "s"(g0), "s"(g1) : "memory");
}

// Per-lane async 16B global -> LDS copy (ASYNCcnt), bypasses VGPRs.
__device__ __forceinline__ void async_b128(uint32_t lds, uint64_t gaddr) {
  asm volatile("global_load_async_to_lds_b128 %0, %1, off"
               :: "v"(lds), "v"(gaddr) : "memory");
}
__device__ __forceinline__ void wait_asynccnt0() {
  asm volatile("s_wait_asynccnt 0x0" ::: "memory");
}

// ---------------------------------------------------------------------------
// 1) LayerNorm (f32 in) -> f16 normalized activations
// ---------------------------------------------------------------------------
__global__ __launch_bounds__(256) void ln_kernel(const float* __restrict__ x,
                                                 const float* __restrict__ w,
                                                 const float* __restrict__ b,
                                                 half_t* __restrict__ xn) {
  __shared__ float red[256];
  const int row = blockIdx.x;
  const int tid = threadIdx.x;
  const size_t base = (size_t)row * DIM + tid * 4;
  const float4 xv = *(const float4*)&x[base];

  red[tid] = xv.x + xv.y + xv.z + xv.w;
  __syncthreads();
  for (int off = 128; off > 0; off >>= 1) {
    if (tid < off) red[tid] += red[tid + off];
    __syncthreads();
  }
  const float mu = red[0] * (1.0f / DIM);
  __syncthreads();

  const float d0 = xv.x - mu, d1 = xv.y - mu, d2 = xv.z - mu, d3 = xv.w - mu;
  red[tid] = d0 * d0 + d1 * d1 + d2 * d2 + d3 * d3;
  __syncthreads();
  for (int off = 128; off > 0; off >>= 1) {
    if (tid < off) red[tid] += red[tid + off];
    __syncthreads();
  }
  const float inv = rsqrtf(red[0] * (1.0f / DIM) + LN_EPS);

  const float4 wv = *(const float4*)&w[tid * 4];
  const float4 bv = *(const float4*)&b[tid * 4];
  xn[base + 0] = (half_t)(d0 * inv * wv.x + bv.x);
  xn[base + 1] = (half_t)(d1 * inv * wv.y + bv.y);
  xn[base + 2] = (half_t)(d2 * inv * wv.z + bv.z);
  xn[base + 3] = (half_t)(d3 * inv * wv.w + bv.w);
}

// ---------------------------------------------------------------------------
// 2) f32 -> f16 conversion (weights)
// ---------------------------------------------------------------------------
__global__ __launch_bounds__(256) void cvt_kernel(const float* __restrict__ in,
                                                  half_t* __restrict__ out, int n) {
  int i = blockIdx.x * 256 + threadIdx.x;
  if (i < n) out[i] = (half_t)in[i];
}

// ---------------------------------------------------------------------------
// 3) WMMA GEMM: C[M,N] (f32) = A[M,K](f16 rm) * B[K,N](f16 rm) [+ bias]
//    128 threads = 4 waves; block tile 64x64; wave tile 32x32 (2x2 frags)
//    A tile staged via async global->LDS; B tile transposed via registers.
// ---------------------------------------------------------------------------
__global__ __launch_bounds__(128) void gemm_kernel(const half_t* __restrict__ A,
                                                   const half_t* __restrict__ Bm,
                                                   const float* __restrict__ bias,
                                                   float* __restrict__ C,
                                                   int M, int Nn, int Kk) {
  __shared__ half_t As[64][40];    // [row][k], padded (80B stride, 16B aligned)
  __shared__ half_t Bst[64][40];   // [col][k], padded

  const int tid  = threadIdx.x;
  const int lane = tid & 31;
  const int w    = tid >> 5;
  const int wm   = w >> 1, wn = w & 1;
  const int row0 = blockIdx.y * 64;
  const int col0 = blockIdx.x * 64;

  v8f acc[2][2];
#pragma unroll
  for (int a = 0; a < 2; ++a)
#pragma unroll
    for (int bnb = 0; bnb < 2; ++bnb)
#pragma unroll
      for (int i = 0; i < 8; ++i) acc[a][bnb][i] = 0.0f;

  const int lc   = lane & 15;
  const int bk   = (lane >> 4) << 3;   // 0 or 8  (A frag K base)
  const int kb   = (lane >> 4) << 4;   // 0 or 16 (unused; B frags also use bk)
  const int rofs = (lane >> 4) << 3;   // C row offset

  for (int k0 = 0; k0 < Kk; k0 += 32) {
    __syncthreads();
#pragma unroll
    for (int it = 0; it < 2; ++it) {
      const int vi = tid + 128 * it;
      {  // A tile 64x32: per-lane async 16B copies, no VGPR staging
        const int r = vi >> 2, c8 = (vi & 3) << 3;
        async_b128(lds_off(&As[r][c8]),
                   (uint64_t)(uintptr_t)&A[(size_t)(row0 + r) * Kk + k0 + c8]);
      }
      {  // B tile 32x64 -> Bst[col][k] (transpose via registers)
        const int r = vi >> 3, c8 = (vi & 7) << 3;
        const v8h g = *(const v8h*)&Bm[(size_t)(k0 + r) * Nn + col0 + c8];
#pragma unroll
        for (int j = 0; j < 8; ++j) Bst[c8 + j][r] = g[j];
      }
    }
    if (k0 + 32 < Kk) {  // speculative prefetch of next tiles
      __builtin_prefetch(&A[(size_t)(row0 + (tid >> 1)) * Kk + k0 + 32], 0, 0);
      __builtin_prefetch(&Bm[(size_t)(k0 + 32 + (tid >> 2)) * Nn + col0], 0, 0);
    }
    wait_asynccnt0();
    __syncthreads();

    Frag16 af[2];
#pragma unroll
    for (int ma = 0; ma < 2; ++ma) {
      const int r = 32 * wm + 16 * ma + lc;
      af[ma].o[0] = *(const v8h*)&As[r][bk];
      af[ma].o[1] = *(const v8h*)&As[r][bk + 16];
    }
#pragma unroll
    for (int nb = 0; nb < 2; ++nb) {
      Frag16 bf;
      const int c = 32 * wn + 16 * nb + lc;
      bf.o[0] = *(const v8h*)&Bst[c][bk];
      bf.o[1] = *(const v8h*)&Bst[c][bk + 16];
#pragma unroll
      for (int ma = 0; ma < 2; ++ma)
        acc[ma][nb] = wmma_f16(af[ma].v, bf.v, acc[ma][nb]);
    }
  }

#pragma unroll
  for (int ma = 0; ma < 2; ++ma)
#pragma unroll
    for (int nb = 0; nb < 2; ++nb) {
      const int col = col0 + 32 * wn + 16 * nb + lc;
      const float bv = bias ? bias[col] : 0.0f;
#pragma unroll
      for (int i = 0; i < 8; ++i) {
        const int r = row0 + 32 * wm + 16 * ma + rofs + i;
        C[(size_t)r * Nn + col] = acc[ma][nb][i] + bv;
      }
    }
}

// ---------------------------------------------------------------------------
// 4) RoPE + head split: qkv f32 (ROWS x 3072) -> per-head f16 Q/K/V [b][h][n][d]
// ---------------------------------------------------------------------------
__global__ __launch_bounds__(256) void rope_kernel(const float* __restrict__ qkv,
                                                   half_t* __restrict__ Qh,
                                                   half_t* __restrict__ Kh,
                                                   half_t* __restrict__ Vh) {
  const int rowg = blockIdx.x;         // b*N + n
  const int n    = rowg & (N_SEQ - 1);
  const int bb   = rowg / N_SEQ;
  const int tid  = threadIdx.x;
  const size_t rbase = (size_t)rowg * QKV3;

  for (int p = tid; p < HEADS * (DH / 2); p += 256) {   // 512 rotation pairs
    const int h  = p >> 5;
    const int dp = p & 31;
    const float ifreq = __powf(10000.0f, -(float)(2 * dp) / (float)DH);
    const float ang = (float)n * ifreq;
    const float c = __cosf(ang), s = __sinf(ang);
    const size_t qo = rbase + h * DH + 2 * dp;
    const float q0 = qkv[qo],          q1 = qkv[qo + 1];
    const float k0 = qkv[qo + INNER],  k1 = qkv[qo + INNER + 1];
    const size_t ob = ((size_t)(bb * HEADS + h) * N_SEQ + n) * DH + 2 * dp;
    Qh[ob]     = (half_t)(q0 * c - q1 * s);
    Qh[ob + 1] = (half_t)(q1 * c + q0 * s);
    Kh[ob]     = (half_t)(k0 * c - k1 * s);
    Kh[ob + 1] = (half_t)(k1 * c + k0 * s);
  }
  for (int i = tid; i < INNER; i += 256) {
    const int h = i >> 6, d = i & 63;
    const size_t ob = ((size_t)(bb * HEADS + h) * N_SEQ + n) * DH + d;
    Vh[ob] = (half_t)qkv[rbase + 2 * INNER + i];
  }
}

// ---------------------------------------------------------------------------
// 5) Flash attention per (b,h): 64 query rows per block, online softmax over
//    64-wide key chunks. Q tile + K chunks staged by the Tensor Data Mover.
// ---------------------------------------------------------------------------
#define CHUNK 64

__global__ __launch_bounds__(128) void attn_kernel(const half_t* __restrict__ Q,
                                                   const half_t* __restrict__ K,
                                                   const half_t* __restrict__ V,
                                                   half_t* __restrict__ Oout) {
  __shared__ half_t Qs[64][72];          // [qrow][d]   (TDM pads 64->72)
  __shared__ half_t Ks[CHUNK][72];       // [key][d]    (TDM pads 64->72)
  __shared__ half_t Vt[DH][CHUNK + 8];   // [d][key]    B-operand layout for P*V
  __shared__ float  Ss[64][CHUNK + 4];   // scores chunk (f32)
  __shared__ half_t Ps[64][72];          // exp(scores) (f16, A operand)
  __shared__ float  rowscale[64];
  __shared__ float  lbuf[64];

  const int tid  = threadIdx.x;
  const int lane = tid & 31;
  const int w    = tid >> 5;
  const int bh   = blockIdx.y;                    // b*HEADS + h
  const int qi0  = blockIdx.x * 64;
  const size_t base = (size_t)bh * N_SEQ * DH;

  const int lc   = lane & 15;
  const int bk   = (lane >> 4) << 3;    // A frag K base (0/8)
  const int kb   = (lane >> 4) << 4;    // unused (0/16); B frags also use bk
  const int rofs = (lane >> 4) << 3;    // C/D row offset (0/8)
  const float scale = 0.125f;           // DH^-0.5

  // Q tile (64x64) via TDM, one wave issues; completion folded into the
  // first chunk's s_wait_tensorcnt.
  if (w == 0) {
    tdm_load_rows_f16(lds_off(&Qs[0][0]),
                      (uint64_t)(uintptr_t)&Q[base + (size_t)qi0 * DH],
                      N_SEQ, 64);
  }

  float m_r = -1e30f, l_r = 0.0f;       // owned by threads 0..63 (row = tid)
  v8f o[4];
#pragma unroll
  for (int nb = 0; nb < 4; ++nb)
#pragma unroll
    for (int i = 0; i < 8; ++i) o[nb][i] = 0.0f;

  for (int j0 = 0; j0 < N_SEQ; j0 += CHUNK) {
    __syncthreads();   // previous chunk's consumers done before restaging

    // K chunk (64x64, contiguous rows) via TDM from wave 0
    if (w == 0) {
      tdm_load_rows_f16(lds_off(&Ks[0][0]),
                        (uint64_t)(uintptr_t)&K[base + (size_t)j0 * DH],
                        N_SEQ, CHUNK);
    }
    // V chunk transposed into Vt (register scatter; needs transpose anyway)
#pragma unroll
    for (int it = 0; it < 4; ++it) {
      const int vi = tid + 128 * it;
      const int r = vi >> 3, c8 = (vi & 7) << 3;
      const v8h g = *(const v8h*)&V[base + (size_t)(j0 + r) * DH + c8];
#pragma unroll
      for (int j = 0; j < 8; ++j) Vt[c8 + j][r] = g[j];
    }
    if (j0 + CHUNK < N_SEQ) {
      __builtin_prefetch(&V[base + (size_t)(j0 + CHUNK + (tid >> 1)) * DH], 0, 0);
    }
    if (w == 0) __builtin_amdgcn_s_wait_tensorcnt(0);
    __syncthreads();

    // S = Q * K^T : wave w -> rows [16w,16w+16) x 64 key cols
    v8f s[4];
#pragma unroll
    for (int nb = 0; nb < 4; ++nb)
#pragma unroll
      for (int i = 0; i < 8; ++i) s[nb][i] = 0.0f;

#pragma unroll
    for (int kk = 0; kk < DH; kk += 32) {
      Frag16 af;
      const int ar = 16 * w + lc;
      af.o[0] = *(const v8h*)&Qs[ar][kk + bk];
      af.o[1] = *(const v8h*)&Qs[ar][kk + bk + 16];
#pragma unroll
      for (int nb = 0; nb < 4; ++nb) {
        Frag16 bf;
        const int kc = 16 * nb + lc;
        bf.o[0] = *(const v8h*)&Ks[kc][kk + bk];
        bf.o[1] = *(const v8h*)&Ks[kc][kk + bk + 16];
        s[nb] = wmma_f16(af.v, bf.v, s[nb]);
      }
    }
#pragma unroll
    for (int nb = 0; nb < 4; ++nb)
#pragma unroll
      for (int i = 0; i < 8; ++i)
        Ss[16 * w + rofs + i][16 * nb + lc] = s[nb][i] * scale;
    __syncthreads();

    // online softmax: thread r (< 64) owns row r of the chunk
    if (tid < 64) {
      float mc = -1e30f;
      for (int c = 0; c < CHUNK; ++c) mc = fmaxf(mc, Ss[tid][c]);
      const float mnew = fmaxf(m_r, mc);
      const float rs   = __expf(m_r - mnew);
      float sum = 0.0f;
      for (int c = 0; c < CHUNK; ++c) {
        const float p = __expf(Ss[tid][c] - mnew);
        Ps[tid][c] = (half_t)p;
        sum += p;
      }
      l_r = l_r * rs + sum;
      m_r = mnew;
      rowscale[tid] = rs;
    }
    __syncthreads();

    // rescale running output, then O += P * V
    float rs[8];
#pragma unroll
    for (int i = 0; i < 8; ++i) rs[i] = rowscale[16 * w + rofs + i];
#pragma unroll
    for (int nb = 0; nb < 4; ++nb)
#pragma unroll
      for (int i = 0; i < 8; ++i) o[nb][i] *= rs[i];

#pragma unroll
    for (int kk = 0; kk < CHUNK; kk += 32) {
      Frag16 af;
      const int ar = 16 * w + lc;
      af.o[0] = *(const v8h*)&Ps[ar][kk + bk];
      af.o[1] = *(const v8h*)&Ps[ar][kk + bk + 16];
#pragma unroll
      for (int nb = 0; nb < 4; ++nb) {
        Frag16 bf;
        const int dc = 16 * nb + lc;
        bf.o[0] = *(const v8h*)&Vt[dc][kk + bk];
        bf.o[1] = *(const v8h*)&Vt[dc][kk + bk + 16];
        o[nb] = wmma_f16(af.v, bf.v, o[nb]);
      }
    }
  }

  if (tid < 64) lbuf[tid] = l_r;
  __syncthreads();

  const int bb = bh / HEADS, h = bh % HEADS;
  float linv[8];
#pragma unroll
  for (int i = 0; i < 8; ++i) linv[i] = 1.0f / lbuf[16 * w + rofs + i];
#pragma unroll
  for (int nb = 0; nb < 4; ++nb)
#pragma unroll
    for (int i = 0; i < 8; ++i) {
      const int r = 16 * w + rofs + i;
      const int col = 16 * nb + lc;
      Oout[((size_t)(bb * N_SEQ) + qi0 + r) * INNER + h * DH + col] =
          (half_t)(o[nb][i] * linv[i]);
    }
}

// ---------------------------------------------------------------------------
// launch
// ---------------------------------------------------------------------------
extern "C" void kernel_launch(void* const* d_in, const int* in_sizes, int n_in,
                              void* d_out, int out_size, void* d_ws, size_t ws_size,
                              hipStream_t stream) {
  (void)in_sizes; (void)n_in; (void)out_size; (void)ws_size;
  const float* x     = (const float*)d_in[0];
  const float* ln_w  = (const float*)d_in[1];
  const float* ln_b  = (const float*)d_in[2];
  const float* w_qkv = (const float*)d_in[3];
  const float* w_out = (const float*)d_in[4];
  const float* b_out = (const float*)d_in[5];
  float* out = (float*)d_out;

  char* ws = (char*)d_ws;
  half_t* xn     = (half_t*)(ws);                                   // 8 MB
  half_t* wqkv_h = (half_t*)(ws + (size_t)8 * 1024 * 1024);         // 6 MB
  half_t* wout_h = (half_t*)(ws + (size_t)14 * 1024 * 1024 + 680 * 1024);
  float*  qkv    = (float*) (ws + (size_t)16 * 1024 * 1024);        // 48 MB
  half_t* qh     = (half_t*)(ws + (size_t)64 * 1024 * 1024);        // 8 MB
  half_t* kh     = (half_t*)(ws + (size_t)72 * 1024 * 1024);        // 8 MB
  half_t* vh     = (half_t*)(ws + (size_t)80 * 1024 * 1024);        // 8 MB
  half_t* attn_h = (half_t*)(ws + (size_t)88 * 1024 * 1024);        // 8 MB

  ln_kernel<<<ROWS, 256, 0, stream>>>(x, ln_w, ln_b, xn);
  cvt_kernel<<<(DIM * QKV3 + 255) / 256, 256, 0, stream>>>(w_qkv, wqkv_h, DIM * QKV3);
  cvt_kernel<<<(INNER * DIM + 255) / 256, 256, 0, stream>>>(w_out, wout_h, INNER * DIM);
  gemm_kernel<<<dim3(QKV3 / 64, ROWS / 64), 128, 0, stream>>>(
      xn, wqkv_h, nullptr, qkv, ROWS, QKV3, DIM);
  rope_kernel<<<ROWS, 256, 0, stream>>>(qkv, qh, kh, vh);
  attn_kernel<<<dim3(N_SEQ / 64, BATCH * HEADS), 128, 0, stream>>>(qh, kh, vh, attn_h);
  gemm_kernel<<<dim3(DIM / 64, ROWS / 64), 128, 0, stream>>>(
      attn_h, wout_h, b_out, out, ROWS, DIM, INNER);
}
